// GNNSet2Set_807453851814
// MI455X (gfx1250) — compile-verified, matrix operand order checked by reading, not run on hardware
//
#include <hip/hip_runtime.h>
#include <hip/hip_bf16.h>

#define N_NODES 50000
#define N_EDGES 1600000
#define NUM_GRAPHS 512
#define HID 128
#define NCLS 10

typedef __attribute__((ext_vector_type(16))) __bf16 v16bf;
typedef __attribute__((ext_vector_type(8)))  float  v8f;

union BFrag { v16bf v; unsigned short u[16]; uint4 q[2]; };

__device__ __forceinline__ unsigned short f2bf(float f) {
  unsigned int u = __float_as_uint(f);
  u += 0x7FFFu + ((u >> 16) & 1u);          // round-to-nearest-even
  return (unsigned short)(u >> 16);
}

// ---- f32 -> bf16 streaming convert (float4 in, ushort4 out) ------------------
__global__ void cvt_bf16x4(const float4* __restrict__ in, ushort4* __restrict__ out, int n4) {
  int i = blockIdx.x * 256 + threadIdx.x;
  if (i >= n4) return;
  float4 v = in[i];
  ushort4 o;
  o.x = f2bf(v.x); o.y = f2bf(v.y); o.z = f2bf(v.z); o.w = f2bf(v.w);
  out[i] = o;
}

// ---- dense GEMM: C[M,128] = A[M,128](bf16) @ W[128,128], bf16 WMMA -----------
// B pre-swizzled in LDS to per-lane fragment layout -> 2x ds_load_b128/frag.
// A is pre-converted bf16 row-major -> 2x global_load_b128/frag (no VALU cvt).
// Each wave owns two 16-row M tiles (block = 8 waves = 256 rows).
__global__ __launch_bounds__(256) void gemm_bf16_wmma(
    const unsigned short* __restrict__ A16, const float* __restrict__ W,
    float* __restrict__ C, int M) {
  __shared__ unsigned short Bs[4 * 8 * 32 * 16];   // 32 KB
  int tid = threadIdx.x;
  // fill: chunk c = {k[2] n[3] lane[5] half[1]} -> 8 consecutive u16 (16B store)
  for (int c = tid; c < 2048; c += 256) {
    int half = c & 1;
    int lane = (c >> 1) & 31;
    int n = (c >> 6) & 7;
    int k = c >> 9;
    int col = n * 16 + (lane & 15);
    int Kst = k * 32 + ((lane >> 4) << 3) + (half << 4);  // contiguous K run of 8
    union { unsigned short u[8]; uint4 q; } t;
#pragma unroll
    for (int e = 0; e < 8; ++e)
      t.u[e] = f2bf(W[(Kst + e) * HID + col]);
    ((uint4*)Bs)[c] = t.q;
  }
  __syncthreads();

  int wave = tid >> 5;
  int lane = tid & 31;
  int laneLo = lane & 15;
  int hi = lane >> 4;
  int rowBase = (blockIdx.x * 8 + wave) * 32;
  int r0 = rowBase + laneLo;       if (r0 >= M) r0 = M - 1;
  int r1 = rowBase + 16 + laneLo;  if (r1 >= M) r1 = M - 1;
  const unsigned short* A0 = A16 + (size_t)r0 * HID;
  const unsigned short* A1 = A16 + (size_t)r1 * HID;

  v8f acc0[8] = {};
  v8f acc1[8] = {};
#pragma unroll
  for (int k = 0; k < 4; ++k) {
    int k0 = k * 32 + hi * 8;                     // 16B aligned
    BFrag a0, a1;
    a0.q[0] = *(const uint4*)(A0 + k0);
    a0.q[1] = *(const uint4*)(A0 + k0 + 16);
    a1.q[0] = *(const uint4*)(A1 + k0);
    a1.q[1] = *(const uint4*)(A1 + k0 + 16);
    const uint4* bq = (const uint4*)Bs + (size_t)(k * 8 * 32 + lane) * 2;
#pragma unroll
    for (int n = 0; n < 8; ++n) {
      BFrag b;
      b.q[0] = bq[0];
      b.q[1] = bq[1];
      bq += 64;                                  // next n-tile (32 lanes * 2 uint4)
      acc0[n] = __builtin_amdgcn_wmma_f32_16x16x32_bf16(
          false, a0.v, false, b.v, (short)0, acc0[n], false, false);
      acc1[n] = __builtin_amdgcn_wmma_f32_16x16x32_bf16(
          false, a1.v, false, b.v, (short)0, acc1[n], false, false);
    }
  }
  // C/D layout: lane<16 -> M=v, lane>=16 -> M=v+8; N = lane&15
#pragma unroll
  for (int n = 0; n < 8; ++n) {
    int col = n * 16 + laneLo;
#pragma unroll
    for (int v = 0; v < 8; ++v) {
      int row0 = rowBase + hi * 8 + v;
      int row1 = row0 + 16;
      if (row0 < M) C[(size_t)row0 * HID + col] = acc0[n][v];
      if (row1 < M) C[(size_t)row1 * HID + col] = acc1[n][v];
    }
  }
}

// ---------------- graph degree / normalization --------------------------------
__global__ void fill_f32(float* __restrict__ p, float val, int n) {
  int i = blockIdx.x * 256 + threadIdx.x;
  if (i < n) p[i] = val;
}
__global__ void deg_accum(const int* __restrict__ dst, float* __restrict__ deg, int E) {
  int i = blockIdx.x * 256 + threadIdx.x;
  if (i < E) atomicAdd(&deg[dst[i]], 1.0f);
}
__global__ void to_dinv(float* __restrict__ deg, int n) {
  int i = blockIdx.x * 256 + threadIdx.x;
  if (i < n) deg[i] = rsqrtf(deg[i]);       // deg >= 1 always (self loop)
}

// ---------------- edge aggregation: one wave per edge, float4 per lane --------
__global__ __launch_bounds__(256) void aggregate_edges(
    const float* __restrict__ h, const int* __restrict__ src,
    const int* __restrict__ dst, const float* __restrict__ dinv,
    float* __restrict__ agg, int E) {
  int e = blockIdx.x * 8 + (threadIdx.x >> 5);
  if (e >= E) return;                        // uniform per wave
  int lane = threadIdx.x & 31;
  int s = src[e], d = dst[e];
  float norm = dinv[s] * dinv[d];
  const float* hr = h + (size_t)s * HID;
  __builtin_prefetch(hr, 0, 0);              // global_prefetch_b8
  float4 v = *((const float4*)hr + lane);
  float* o = agg + (size_t)d * HID + lane * 4;
  atomicAdd(o + 0, norm * v.x);
  atomicAdd(o + 1, norm * v.y);
  atomicAdd(o + 2, norm * v.z);
  atomicAdd(o + 3, norm * v.w);
}

// -------- self-loop + bias + ReLU; also emits bf16 copy for next GEMM ---------
__global__ void bias_relu_self(const float* __restrict__ agg,
                               const float* __restrict__ xw,
                               const float* __restrict__ dinv,
                               const float* __restrict__ b,
                               float* __restrict__ out,
                               unsigned short* __restrict__ out16, int n) {
  int i = blockIdx.x * 256 + threadIdx.x;
  if (i >= n * HID) return;
  int node = i >> 7, f = i & 127;
  float di = dinv[node];
  float v = agg[i] + di * di * xw[i] + b[f];
  v = v > 0.f ? v : 0.f;
  out[i] = v;
  out16[i] = f2bf(v);
}

// ---------------- Set2Set -----------------------------------------------------
__global__ void zero3(float* a, float* b, float* c, int n) {
  int i = blockIdx.x * 256 + threadIdx.x;
  if (i < n) { a[i] = 0.f; b[i] = 0.f; c[i] = 0.f; }
}

__global__ __launch_bounds__(256) void lstm_gates(
    const float* __restrict__ hl, const float* __restrict__ r,
    const float* __restrict__ Wih, const float* __restrict__ Whh,
    const float* __restrict__ bih, const float* __restrict__ bhh,
    float* __restrict__ gates) {
  int idx = blockIdx.x * 256 + threadIdx.x;  // b*512 + j
  if (idx >= NUM_GRAPHS * 4 * HID) return;
  int b = idx >> 9, j = idx & 511;
  const float4* hv = (const float4*)(hl + b * HID);
  const float4* rv = (const float4*)(r + b * HID);
  const float4* wi = (const float4*)(Wih + (size_t)j * 2 * HID);
  const float4* wh = (const float4*)(Whh + (size_t)j * HID);
  float s = bih[j] + bhh[j];
#pragma unroll 4
  for (int t = 0; t < HID / 4; ++t) {
    float4 h4 = hv[t], r4 = rv[t];
    float4 wiq = wi[t], wir = wi[t + HID / 4], whq = wh[t];
    s += h4.x * (wiq.x + whq.x) + h4.y * (wiq.y + whq.y) +
         h4.z * (wiq.z + whq.z) + h4.w * (wiq.w + whq.w);
    s += r4.x * wir.x + r4.y * wir.y + r4.z * wir.z + r4.w * wir.w;
  }
  gates[idx] = s;
}

__device__ __forceinline__ float sigm(float x) { return 1.f / (1.f + expf(-x)); }

__global__ void lstm_update(const float* __restrict__ gates,
                            float* __restrict__ c, float* __restrict__ h,
                            float* __restrict__ r,
                            unsigned int* __restrict__ mkey,
                            float* __restrict__ denom) {
  int idx = blockIdx.x * 256 + threadIdx.x;  // b*128 + k
  if (idx >= NUM_GRAPHS * HID) return;
  int b = idx >> 7, k = idx & 127;
  const float* g = gates + (size_t)b * 4 * HID;
  float gi = sigm(g[k]), gf = sigm(g[HID + k]);
  float gg = tanhf(g[2 * HID + k]), go = sigm(g[3 * HID + k]);
  float cn = gf * c[idx] + gi * gg;
  c[idx] = cn;
  h[idx] = go * tanhf(cn);
  r[idx] = 0.f;                               // fresh accumulator for this step
  if (k == 0) { mkey[b] = 0u; denom[b] = 0.f; }
}

__device__ __forceinline__ unsigned int fkey(float f) {
  unsigned int s = __float_as_uint(f);
  return (s & 0x80000000u) ? ~s : (s | 0x80000000u);  // order-preserving map
}
__device__ __forceinline__ float fdec(unsigned int k) {
  if (k == 0u) return 0.f;                    // empty segment -> m = 0 (ref isfinite path)
  unsigned int s = (k & 0x80000000u) ? (k & 0x7FFFFFFFu) : ~k;
  return __uint_as_float(s);
}

__global__ void attn_e(const float* __restrict__ hfeat, const float* __restrict__ hl,
                       const int* __restrict__ batch, float* __restrict__ ebuf,
                       unsigned int* __restrict__ mkey) {
  int n = blockIdx.x * 256 + threadIdx.x;
  if (n >= N_NODES) return;
  int b = batch[n];
  const float4* x4 = (const float4*)(hfeat + (size_t)n * HID);
  const float4* q4 = (const float4*)(hl + (size_t)b * HID);
  float s = 0.f;
#pragma unroll 8
  for (int t = 0; t < HID / 4; ++t) {
    float4 a = x4[t], q = q4[t];
    s += a.x * q.x + a.y * q.y + a.z * q.z + a.w * q.w;
  }
  ebuf[n] = s;
  atomicMax(mkey + b, fkey(s));
}

__global__ void attn_z(const float* __restrict__ ebuf, const int* __restrict__ batch,
                       const unsigned int* __restrict__ mkey,
                       float* __restrict__ zbuf, float* __restrict__ denom) {
  int n = blockIdx.x * 256 + threadIdx.x;
  if (n >= N_NODES) return;
  int b = batch[n];
  float z = expf(ebuf[n] - fdec(mkey[b]));
  zbuf[n] = z;
  atomicAdd(denom + b, z);
}

__global__ void attn_r(const float* __restrict__ hfeat, const float* __restrict__ zbuf,
                       const float* __restrict__ denom, const int* __restrict__ batch,
                       float* __restrict__ r) {
  int idx = blockIdx.x * 256 + threadIdx.x;  // n*32 + t
  if (idx >= N_NODES * 32) return;
  int n = idx >> 5, t = idx & 31;
  int b = batch[n];
  float a = zbuf[n] / denom[b];
  float4 v = *((const float4*)(hfeat + (size_t)n * HID) + t);
  float* o = r + (size_t)b * HID + t * 4;
  atomicAdd(o + 0, a * v.x);
  atomicAdd(o + 1, a * v.y);
  atomicAdd(o + 2, a * v.z);
  atomicAdd(o + 3, a * v.w);
}

// ---------------- final MLP ---------------------------------------------------
__global__ void mlp1(const float* __restrict__ hl, const float* __restrict__ r,
                     const float* __restrict__ Wl1, const float* __restrict__ bl1,
                     float* __restrict__ t) {
  int idx = blockIdx.x * 256 + threadIdx.x;  // b*128 + j
  if (idx >= NUM_GRAPHS * HID) return;
  int b = idx >> 7, j = idx & 127;
  const float* hq = hl + (size_t)b * HID;
  const float* rq = r + (size_t)b * HID;
  float s = bl1[j];
#pragma unroll 4
  for (int k = 0; k < HID; ++k)
    s += hq[k] * Wl1[k * HID + j] + rq[k] * Wl1[(HID + k) * HID + j];
  t[idx] = s > 0.f ? s : 0.f;
}

__global__ void mlp2(const float* __restrict__ t, const float* __restrict__ Wl2,
                     const float* __restrict__ bl2, float* __restrict__ out) {
  int idx = blockIdx.x * 256 + threadIdx.x;  // b*10 + c
  if (idx >= NUM_GRAPHS * NCLS) return;
  int b = idx / NCLS, c = idx % NCLS;
  const float* tq = t + (size_t)b * HID;
  float s = bl2[c];
#pragma unroll 4
  for (int j = 0; j < HID; ++j) s += tq[j] * Wl2[j * NCLS + c];
  out[idx] = s;
}

// ---------------- launch ------------------------------------------------------
extern "C" void kernel_launch(void* const* d_in, const int* in_sizes, int n_in,
                              void* d_out, int out_size, void* d_ws, size_t ws_size,
                              hipStream_t stream) {
  (void)in_sizes; (void)n_in; (void)out_size; (void)ws_size;
  const float* x    = (const float*)d_in[0];
  const int*   ei   = (const int*)d_in[1];
  const int*   batch= (const int*)d_in[2];
  const float* W1   = (const float*)d_in[3];
  const float* b1   = (const float*)d_in[4];
  const float* W2   = (const float*)d_in[5];
  const float* b2   = (const float*)d_in[6];
  const float* Wih  = (const float*)d_in[7];
  const float* Whh  = (const float*)d_in[8];
  const float* bih  = (const float*)d_in[9];
  const float* bhh  = (const float*)d_in[10];
  const float* Wl1  = (const float*)d_in[11];
  const float* bl1  = (const float*)d_in[12];
  const float* Wl2  = (const float*)d_in[13];
  const float* bl2  = (const float*)d_in[14];
  const int* src = ei;
  const int* dst = ei + N_EDGES;
  float* out = (float*)d_out;

  char* base = (char*)d_ws;
  size_t off = 0;
  auto alloc = [&](size_t bytes) -> void* {
    void* q = base + off;
    off += (bytes + 255) & ~(size_t)255;
    return q;
  };
  float* dinv  = (float*)alloc((size_t)N_NODES * 4);          // deg -> dinv in place
  float* xw    = (float*)alloc((size_t)N_NODES * HID * 4);
  float* agg   = (float*)alloc((size_t)N_NODES * HID * 4);
  float* h     = (float*)alloc((size_t)N_NODES * HID * 4);
  unsigned short* abf = (unsigned short*)alloc((size_t)N_NODES * HID * 2); // bf16 activations
  float* gates = (float*)alloc((size_t)NUM_GRAPHS * 4 * HID * 4);
  float* hl    = (float*)alloc((size_t)NUM_GRAPHS * HID * 4);
  float* cl    = (float*)alloc((size_t)NUM_GRAPHS * HID * 4);
  float* rl    = (float*)alloc((size_t)NUM_GRAPHS * HID * 4);
  float* ebuf  = (float*)alloc((size_t)N_NODES * 4);
  float* zbuf  = (float*)alloc((size_t)N_NODES * 4);
  unsigned int* mkey = (unsigned int*)alloc((size_t)NUM_GRAPHS * 4);
  float* denom = (float*)alloc((size_t)NUM_GRAPHS * 4);
  float* tbuf  = (float*)alloc((size_t)NUM_GRAPHS * HID * 4);

  // degree / symmetric norm (shared by both layers)
  fill_f32<<<(N_NODES + 255) / 256, 256, 0, stream>>>(dinv, 1.0f, N_NODES);
  deg_accum<<<(N_EDGES + 255) / 256, 256, 0, stream>>>(dst, dinv, N_EDGES);
  to_dinv<<<(N_NODES + 255) / 256, 256, 0, stream>>>(dinv, N_NODES);

  // x -> bf16 once; layer-2 bf16 copy is fused into bias_relu_self
  cvt_bf16x4<<<(N_NODES * HID / 4 + 255) / 256, 256, 0, stream>>>(
      (const float4*)x, (ushort4*)abf, N_NODES * HID / 4);

  for (int layer = 0; layer < 2; ++layer) {
    const float* W  = (layer == 0) ? W1 : W2;
    const float* bb = (layer == 0) ? b1 : b2;
    gemm_bf16_wmma<<<(N_NODES + 255) / 256, 256, 0, stream>>>(abf, W, xw, N_NODES);
    fill_f32<<<(N_NODES * HID + 255) / 256, 256, 0, stream>>>(agg, 0.f, N_NODES * HID);
    aggregate_edges<<<(N_EDGES + 7) / 8, 256, 0, stream>>>(xw, src, dst, dinv, agg, N_EDGES);
    bias_relu_self<<<(N_NODES * HID + 255) / 256, 256, 0, stream>>>(
        agg, xw, dinv, bb, h, abf, N_NODES);
  }

  // Set2Set over graphs
  zero3<<<(NUM_GRAPHS * HID + 255) / 256, 256, 0, stream>>>(hl, cl, rl, NUM_GRAPHS * HID);
  for (int step = 0; step < 3; ++step) {
    lstm_gates<<<(NUM_GRAPHS * 4 * HID + 255) / 256, 256, 0, stream>>>(hl, rl, Wih, Whh, bih, bhh, gates);
    lstm_update<<<(NUM_GRAPHS * HID + 255) / 256, 256, 0, stream>>>(gates, cl, hl, rl, mkey, denom);
    attn_e<<<(N_NODES + 255) / 256, 256, 0, stream>>>(h, hl, batch, ebuf, mkey);
    attn_z<<<(N_NODES + 255) / 256, 256, 0, stream>>>(ebuf, batch, mkey, zbuf, denom);
    attn_r<<<(N_NODES * 32 + 255) / 256, 256, 0, stream>>>(h, zbuf, denom, batch, rl);
  }

  // readout MLP
  mlp1<<<(NUM_GRAPHS * HID + 255) / 256, 256, 0, stream>>>(hl, rl, Wl1, bl1, tbuf);
  mlp2<<<(NUM_GRAPHS * NCLS + 255) / 256, 256, 0, stream>>>(tbuf, Wl2, bl2, out);
}
